// STHMLayer_80496277062135
// MI455X (gfx1250) — compile-verified, matrix operand order checked by reading, not run on hardware
//
#include <hip/hip_runtime.h>
#include <hip/hip_bf16.h>
#include <math.h>
#include <stdint.h>

// ---------------------------------------------------------------------------
// CDNA5 (gfx1250) wave32 / WMMA / TDM implementation of the STHM layer fwd.
// ---------------------------------------------------------------------------

typedef __attribute__((ext_vector_type(16))) _Float16 v16h;
typedef __attribute__((ext_vector_type(8)))  float    v8f;
typedef unsigned int uint32x4 __attribute__((ext_vector_type(4)));
typedef int          int32x8  __attribute__((ext_vector_type(8)));
typedef int          int32x4  __attribute__((ext_vector_type(4)));

#if __has_builtin(__builtin_amdgcn_tensor_load_to_lds)
#define USE_TDM 1
#else
#define USE_TDM 0
#endif

#define LDSTR 40   // padded LDS row stride in halfs (32 data + 8 pad = 80B)

static __device__ __forceinline__ void tdm_wait_all() {
#if __has_builtin(__builtin_amdgcn_s_wait_tensorcnt)
    __builtin_amdgcn_s_wait_tensorcnt(0);
#endif
}

// ============================================================================
// GEMM: C[M,N] = act( A[M,K](fp32) @ W[K,N] + bias[N] ), weights pre-packed as
// Wt[N,K] f16.  Block tile 128x128, 256 threads = 8 waves (4M x 2N), each wave
// 32x64 = 2x4 v_wmma_f32_16x16x32_f16.  Double-buffered LDS: A tile loaded to
// registers ahead of time (branch-free, mask-zeroed), B tile fetched by the
// Tensor Data Mover with hardware row padding.
// ============================================================================
__global__ __launch_bounds__(256)
void gemm_wmma_kernel(const float* __restrict__ A, const _Float16* __restrict__ Wt,
                      const float* __restrict__ bias, float* __restrict__ C,
                      int M, int N, int K, int act_gelu)
{
    __shared__ _Float16 ldsA[2][128 * LDSTR];
    __shared__ _Float16 ldsB[2][128 * LDSTR];

    const int t     = threadIdx.x;
    const int lane  = t & 31;
    const int wid   = t >> 5;
    const int waveM = wid >> 1;          // 0..3 -> 32-row strip
    const int waveN = wid & 1;           // 0..1 -> 64-col strip
    const int m0    = blockIdx.y * 128;
    const int n0    = blockIdx.x * 128;
    const int frow  = lane & 15;
    const int ksel  = (lane >> 4) << 3;  // lanes 16..31 take K+8 per ISA layout

    // Per-thread A-tile slots: 4 x float4 covering the 128x32 fp32 tile.
    // Branch-free OOB: clamp row pointer, zero via multiply mask.
    int arow[4], acol[4];
    const float* aptr[4];
    float amask[4];
    #pragma unroll
    for (int i = 0; i < 4; ++i) {
        int f4  = t + i * 256;
        arow[i] = f4 >> 3;
        acol[i] = (f4 & 7) << 2;
        int gm  = m0 + arow[i];
        aptr[i] = A + (size_t)(gm < M ? gm : (M - 1)) * K + acol[i];
        amask[i] = (gm < M) ? 1.0f : 0.0f;
    }

    float4 areg[4];
    auto load_a = [&](int kt) {
        #pragma unroll
        for (int i = 0; i < 4; ++i)
            areg[i] = *(const float4*)(aptr[i] + kt);
    };
    auto store_a = [&](int buf) {
        #pragma unroll
        for (int i = 0; i < 4; ++i) {
            _Float16* d = &ldsA[buf][arow[i] * LDSTR + acol[i]];
            d[0] = (_Float16)(areg[i].x * amask[i]);
            d[1] = (_Float16)(areg[i].y * amask[i]);
            d[2] = (_Float16)(areg[i].z * amask[i]);
            d[3] = (_Float16)(areg[i].w * amask[i]);
        }
    };

#if USE_TDM
    // Tensor Data Mover: 32(K) x 128(N) f16 tile of Wt (N-major), with HW pad
    // (16 DWORDs data + 4 DWORDs pad) -> 80B LDS rows == LDSTR halfs.
    auto fetch_b = [&](int kt, int buf) {
        if (wid != 0) return;
        unsigned long long ga =
            (unsigned long long)(uintptr_t)(Wt + (size_t)n0 * K + kt);
        unsigned lds_off = (unsigned)(uintptr_t)(&ldsB[buf][0]);
        uint32x4 g0;
        g0[0] = 1u;                                            // count=1, user
        g0[1] = lds_off;                                       // lds_addr
        g0[2] = (unsigned)(ga & 0xffffffffull);                // global_addr lo
        g0[3] = (unsigned)((ga >> 32) & 0x01fffffful) | (2u << 30); // hi|type=2
        int32x8 g1;
        g1[0] = (1 << 16) | (1 << 20) | (3 << 22) | (3 << 25); // 2B, pad 16+4dw
        g1[1] = (K & 0xffff) << 16;                            // tensor_dim0 lo
        g1[2] = ((unsigned)K >> 16) | ((N & 0xffff) << 16);    // dim0 hi|dim1 lo
        g1[3] = ((unsigned)N >> 16) | (32 << 16);              // dim1 hi|tile0=32
        g1[4] = 128;                                           // tile1=128,tile2=0
        g1[5] = K;                                             // dim0_stride lo
        g1[6] = 0;
        g1[7] = 0;
        int32x4 z4 = {0, 0, 0, 0};
        int32x8 z8 = {0, 0, 0, 0, 0, 0, 0, 0};
        __builtin_amdgcn_tensor_load_to_lds(g0, g1, z4, z4, z8, 0);
    };
    auto wait_b = [&]() { if (wid == 0) tdm_wait_all(); };
#else
    // Fallback: cooperative copy, 16 halfs per thread.
    auto fetch_b = [&](int kt, int buf) {
        const _Float16* src = Wt + (size_t)(n0 + (t >> 1)) * K + kt + ((t & 1) << 4);
        _Float16* dst = &ldsB[buf][(t >> 1) * LDSTR + ((t & 1) << 4)];
        ((float4*)dst)[0] = ((const float4*)src)[0];
        ((float4*)dst)[1] = ((const float4*)src)[1];
    };
    auto wait_b = [&]() {};
#endif

    v8f acc[2][4];
    {
        v8f z = {};
        #pragma unroll
        for (int i = 0; i < 2; ++i)
            #pragma unroll
            for (int j = 0; j < 4; ++j)
                acc[i][j] = z;
    }

    // ---- prologue: stage tile 0 ----
    load_a(0);
    fetch_b(0, 0);
    store_a(0);
    wait_b();
    __syncthreads();

    int buf = 0;
    for (int kt = 0; kt < K; kt += 32, buf ^= 1) {
        const bool more = (kt + 32) < K;
        if (more) {
            load_a(kt + 32);                 // global loads in flight over WMMAs
            fetch_b(kt + 32, buf ^ 1);       // async TDM into the other buffer
            if (kt + 64 < K)
                __builtin_prefetch(aptr[0] + kt + 64, 0, 1);
        }

        // ---- fragments per ISA 16-bit A/B VGPR layout + 8 WMMAs ----
        v16h afrag[2], bfrag[4];
        #pragma unroll
        for (int mt = 0; mt < 2; ++mt) {
            const _Float16* p = &ldsA[buf][(waveM * 32 + mt * 16 + frow) * LDSTR];
            v16h a;
            #pragma unroll
            for (int v = 0; v < 8; ++v) {
                int kk = ((v & 4) << 2) + ksel + ((v & 3) << 1);
                a[2 * v]     = p[kk];
                a[2 * v + 1] = p[kk + 1];
            }
            afrag[mt] = a;
        }
        #pragma unroll
        for (int nt = 0; nt < 4; ++nt) {
            const _Float16* p = &ldsB[buf][(waveN * 64 + nt * 16 + frow) * LDSTR];
            v16h bf;
            #pragma unroll
            for (int v = 0; v < 8; ++v) {
                int kk = ((v & 4) << 2) + ksel + ((v & 3) << 1);
                bf[2 * v]     = p[kk];
                bf[2 * v + 1] = p[kk + 1];
            }
            bfrag[nt] = bf;
        }
        #pragma unroll
        for (int mt = 0; mt < 2; ++mt)
            #pragma unroll
            for (int nt = 0; nt < 4; ++nt)
                acc[mt][nt] = __builtin_amdgcn_wmma_f32_16x16x32_f16(
                    false, afrag[mt], false, bfrag[nt],
                    (short)0, acc[mt][nt], false, false);

        if (more) {
            store_a(buf ^ 1);                // converts wait on loads here, late
            wait_b();
        }
        __syncthreads();
    }

    // ---- epilogue: (VGPR r, lane l) -> M = 8*(l/16)+r, N = l%16 ----
    const int mrow8 = (lane >> 4) << 3;
    #pragma unroll
    for (int mt = 0; mt < 2; ++mt) {
        #pragma unroll
        for (int nt = 0; nt < 4; ++nt) {
            int gn   = n0 + waveN * 64 + nt * 16 + frow;
            float bv = bias[gn];
            v8f c = acc[mt][nt];
            #pragma unroll
            for (int r = 0; r < 8; ++r) {
                int gm = m0 + waveM * 32 + mt * 16 + mrow8 + r;
                if (gm < M) {
                    float x = c[r] + bv;
                    if (act_gelu)
                        x = 0.5f * x * (1.0f + erff(x * 0.7071067811865476f));
                    C[(size_t)gm * N + gn] = x;
                }
            }
        }
    }
}

// ============================================================================
// Weight pre-pack: W[K,N] fp32 -> Wt[N,K] f16 (32x32 LDS-tiled transpose).
// ============================================================================
__global__ __launch_bounds__(256)
void transpose_convert_kernel(const float* __restrict__ W, _Float16* __restrict__ Wt,
                              int K, int N)
{
    __shared__ float tile[32][33];
    const int kb = blockIdx.y * 32, nb = blockIdx.x * 32;
    const int tx = threadIdx.x & 31, ty = threadIdx.x >> 5;   // 32 x 8
    #pragma unroll
    for (int r = 0; r < 32; r += 8)
        tile[ty + r][tx] = W[(size_t)(kb + ty + r) * N + nb + tx];
    __syncthreads();
    #pragma unroll
    for (int r = 0; r < 32; r += 8)
        Wt[(size_t)(nb + ty + r) * K + kb + tx] = (_Float16)tile[tx][ty + r];
}

// ============================================================================
// Entmax-1.5 attention. One wave per (batch, head, query-row). Output written
// contiguous (B, H, L, 64) to reproduce the reference's mix=True view.
// ============================================================================
__global__ __launch_bounds__(128)
void attn_entmax_kernel(const float* __restrict__ Q, const float* __restrict__ K,
                        const float* __restrict__ V, float* __restrict__ O,
                        int B, int H, int L, int S, long long kv_bstride,
                        float scale)
{
    __shared__ __align__(16) float qs[4][64];
    __shared__ float ps[4][64];

    const int lane = threadIdx.x;   // 0..31
    const int ty   = threadIdx.y;   // 0..3
    const int lb   = L >> 2;
    int idx  = blockIdx.x;
    const int lblk = idx % lb; idx /= lb;
    const int h    = idx % H;
    const int b    = idx / H;
    const int l    = lblk * 4 + ty;

    const float* qrow = Q + ((size_t)(b * L + l)) * 1024 + h * 64;
    qs[ty][lane]      = qrow[lane];
    qs[ty][lane + 32] = qrow[lane + 32];
    __syncthreads();

    const float* kbase = K + (size_t)b * kv_bstride + h * 64;
    const float* vbase = V + (size_t)b * kv_bstride + h * 64;

    float sc[2] = { -1e30f, -1e30f };
    #pragma unroll
    for (int i = 0; i < 2; ++i) {
        int s = lane + i * 32;
        if (s < S) {
            const float4* kr4 = (const float4*)(kbase + (size_t)s * 1024);
            const float4* q4  = (const float4*)qs[ty];
            float a = 0.f;
            #pragma unroll
            for (int e = 0; e < 16; ++e) {
                float4 kv = kr4[e], qv = q4[e];
                a += qv.x * kv.x + qv.y * kv.y + qv.z * kv.z + qv.w * kv.w;
            }
            sc[i] = a * scale * 0.5f;       // xs = (alpha-1) * score
        }
    }

    float mx = fmaxf(sc[0], sc[1]);
    #pragma unroll
    for (int m = 16; m > 0; m >>= 1) mx = fmaxf(mx, __shfl_xor(mx, m, 32));

    float lo = mx - 1.0f;
    float hi = mx - rsqrtf((float)S);
    for (int it = 0; it < 30; ++it) {
        float mid = 0.5f * (lo + hi);
        float f = 0.f;
        #pragma unroll
        for (int i = 0; i < 2; ++i) {
            float d = sc[i] - mid;
            if (d > 0.f) f += d * d;
        }
        #pragma unroll
        for (int m = 16; m > 0; m >>= 1) f += __shfl_xor(f, m, 32);
        if (f - 1.0f > 0.f) lo = mid; else hi = mid;
    }
    float tau = 0.5f * (lo + hi);

    float pv[2], psum = 0.f;
    #pragma unroll
    for (int i = 0; i < 2; ++i) {
        float d = sc[i] - tau;
        pv[i] = (d > 0.f) ? d * d : 0.f;
        psum += pv[i];
    }
    #pragma unroll
    for (int m = 16; m > 0; m >>= 1) psum += __shfl_xor(psum, m, 32);
    float inv = 1.0f / psum;

    #pragma unroll
    for (int i = 0; i < 2; ++i) {
        int s = lane + i * 32;
        if (s < S) ps[ty][s] = pv[i] * inv;
    }
    __syncthreads();

    float o0 = 0.f, o1 = 0.f;
    for (int s = 0; s < S; ++s) {
        float p = ps[ty][s];
        const float* vr = vbase + (size_t)s * 1024;
        o0 += p * vr[lane];
        o1 += p * vr[lane + 32];
    }
    float* out = O + ((size_t)((b * H + h) * L + l)) * 64;
    out[lane]      = o0;
    out[lane + 32] = o1;
}

// ============================================================================
// Residual add + LayerNorm over rows of 1024. One 256-thread block per row.
// ============================================================================
__global__ __launch_bounds__(256)
void add_ln_kernel(const float* __restrict__ X, const float* __restrict__ Y,
                   const float* __restrict__ g, const float* __restrict__ bta,
                   float* __restrict__ Out)
{
    __shared__ float red[256];
    const int t = threadIdx.x;
    const size_t base = (size_t)blockIdx.x * 1024;

    float v[4]; float s = 0.f;
    #pragma unroll
    for (int i = 0; i < 4; ++i) {
        int c = t + i * 256;
        v[i] = X[base + c] + Y[base + c];
        s += v[i];
    }
    red[t] = s; __syncthreads();
    #pragma unroll
    for (int off = 128; off > 0; off >>= 1) {
        if (t < off) red[t] += red[t + off];
        __syncthreads();
    }
    float mean = red[0] * (1.0f / 1024.0f);
    __syncthreads();

    float vs = 0.f;
    #pragma unroll
    for (int i = 0; i < 4; ++i) { float d = v[i] - mean; vs += d * d; }
    red[t] = vs; __syncthreads();
    #pragma unroll
    for (int off = 128; off > 0; off >>= 1) {
        if (t < off) red[t] += red[t + off];
        __syncthreads();
    }
    float inv = rsqrtf(red[0] * (1.0f / 1024.0f) + 1e-5f);

    #pragma unroll
    for (int i = 0; i < 4; ++i) {
        int c = t + i * 256;
        Out[base + c] = (v[i] - mean) * inv * g[c] + bta[c];
    }
}

// ============================================================================
// (B, d1, d2, 1024) -> (B, d2, d1, 1024) middle-dim transpose (row copy).
// ============================================================================
__global__ __launch_bounds__(256)
void permute_mid_kernel(const float* __restrict__ src, float* __restrict__ dst,
                        int d1, int d2)
{
    int r = blockIdx.x;
    int j = r % d2; r /= d2;
    int i = r % d1;
    int b = r / d1;
    const float4* s = (const float4*)(src + ((size_t)((b * d1 + i) * d2 + j)) * 1024);
    float4*       d = (float4*)(dst + ((size_t)((b * d2 + j) * d1 + i)) * 1024);
    d[threadIdx.x] = s[threadIdx.x];
}

// ============================================================================
// Host orchestration.
// ============================================================================
extern "C" void kernel_launch(void* const* d_in, const int* in_sizes, int n_in,
                              void* d_out, int out_size, void* d_ws, size_t ws_size,
                              hipStream_t stream)
{
    (void)in_sizes; (void)n_in; (void)out_size; (void)ws_size;

    const float* x      = (const float*)d_in[0];
    const float* ct_wq  = (const float*)d_in[1];  const float* ct_bq = (const float*)d_in[2];
    const float* ct_wk  = (const float*)d_in[3];  const float* ct_bk = (const float*)d_in[4];
    const float* ct_wv  = (const float*)d_in[5];  const float* ct_bv = (const float*)d_in[6];
    const float* ct_wo  = (const float*)d_in[7];  const float* ct_bo = (const float*)d_in[8];
    const float* cs_wq  = (const float*)d_in[9];  const float* cs_bq = (const float*)d_in[10];
    const float* cs_wk  = (const float*)d_in[11]; const float* cs_bk = (const float*)d_in[12];
    const float* cs_wv  = (const float*)d_in[13]; const float* cs_bv = (const float*)d_in[14];
    const float* cs_wo  = (const float*)d_in[15]; const float* cs_bo = (const float*)d_in[16];
    const float* hp_wq  = (const float*)d_in[17]; const float* hp_bq = (const float*)d_in[18];
    const float* hp_wk  = (const float*)d_in[19]; const float* hp_bk = (const float*)d_in[20];
    const float* hp_wv  = (const float*)d_in[21]; const float* hp_bv = (const float*)d_in[22];
    const float* hp_wo  = (const float*)d_in[23]; const float* hp_bo = (const float*)d_in[24];
    const float* cs_key = (const float*)d_in[25];
    const float* n1_g   = (const float*)d_in[26]; const float* n1_b  = (const float*)d_in[27];
    const float* n2_g   = (const float*)d_in[28]; const float* n2_b  = (const float*)d_in[29];
    const float* n3_g   = (const float*)d_in[30]; const float* n3_b  = (const float*)d_in[31];
    const float* n4_g   = (const float*)d_in[32]; const float* n4_b  = (const float*)d_in[33];
    const float* m1_w1  = (const float*)d_in[34]; const float* m1_b1 = (const float*)d_in[35];
    const float* m1_w2  = (const float*)d_in[36]; const float* m1_b2 = (const float*)d_in[37];
    const float* m2_w1  = (const float*)d_in[38]; const float* m2_b1 = (const float*)d_in[39];
    const float* m2_w2  = (const float*)d_in[40]; const float* m2_b2 = (const float*)d_in[41];

    const int BB = 8, TSD = 32, SEG = 64, D = 1024, H = 16, DFF = 4096, FACTOR = 10;
    const int T  = BB * TSD * SEG;                 // 16384 tokens per stage
    const int B1 = BB * TSD;                       // 256 time batches, L=S=64
    const int B2 = BB * SEG;                       // 512 series batches, L=32
    const float SCALE = 0.125f;                    // 1/sqrt(64)

    float* w = (float*)d_ws;
    const size_t TD = (size_t)T * D;
    float* bufQ    = w; w += TD;
    float* bufK    = w; w += TD;
    float* bufV    = w; w += TD;
    float* bufAttn = w; w += TD;
    float* bufR1   = w; w += TD;
    float* bufR2   = w; w += TD;
    float* bufS    = w; w += TD;
    float* bufA2   = w; w += TD;
    float* bufB2   = w; w += TD;
    float* bufH    = w; w += (size_t)T * DFF;
    float* csK     = w; w += (size_t)FACTOR * D;
    float* csV     = w; w += (size_t)FACTOR * D;

    // f16 transposed weight pool (after the fp32 pool; 16B-aligned offsets).
    _Float16* hw = (_Float16*)w;
    auto alloc_h = [&](size_t n) { _Float16* p = hw; hw += n; return p; };
    const size_t DD = (size_t)D * D, DF = (size_t)D * DFF;
    _Float16* t_ct_wq = alloc_h(DD); _Float16* t_ct_wk = alloc_h(DD);
    _Float16* t_ct_wv = alloc_h(DD); _Float16* t_ct_wo = alloc_h(DD);
    _Float16* t_cs_wq = alloc_h(DD); _Float16* t_cs_wk = alloc_h(DD);
    _Float16* t_cs_wv = alloc_h(DD); _Float16* t_cs_wo = alloc_h(DD);
    _Float16* t_hp_wq = alloc_h(DD); _Float16* t_hp_wk = alloc_h(DD);
    _Float16* t_hp_wv = alloc_h(DD); _Float16* t_hp_wo = alloc_h(DD);
    _Float16* t_m1_w1 = alloc_h(DF); _Float16* t_m1_w2 = alloc_h(DF);
    _Float16* t_m2_w1 = alloc_h(DF); _Float16* t_m2_w2 = alloc_h(DF);

    auto tconv = [&](const float* Ws, _Float16* Wd, int K, int N) {
        transpose_convert_kernel<<<dim3(N / 32, K / 32), 256, 0, stream>>>(Ws, Wd, K, N);
    };
    auto gemm = [&](const float* A, const _Float16* Wth, const float* bias, float* C,
                    int M, int N, int K, int gelu) {
        dim3 grid(N / 128, (M + 127) / 128);
        gemm_wmma_kernel<<<grid, 256, 0, stream>>>(A, Wth, bias, C, M, N, K, gelu);
    };
    auto attn = [&](const float* Qp, const float* Kp, const float* Vp, float* Op,
                    int Bn, int L, int S, long long kvs) {
        attn_entmax_kernel<<<dim3(Bn * H * (L / 4)), dim3(32, 4), 0, stream>>>(
            Qp, Kp, Vp, Op, Bn, H, L, S, kvs, SCALE);
    };
    auto addln = [&](const float* X, const float* Y, const float* g, const float* b,
                     float* Out) {
        add_ln_kernel<<<dim3(T), 256, 0, stream>>>(X, Y, g, b, Out);
    };

    // ---------------- Weight pre-pack (fp32 KxN -> f16 NxK) ----------------
    tconv(ct_wq, t_ct_wq, D, D); tconv(ct_wk, t_ct_wk, D, D);
    tconv(ct_wv, t_ct_wv, D, D); tconv(ct_wo, t_ct_wo, D, D);
    tconv(cs_wq, t_cs_wq, D, D); tconv(cs_wk, t_cs_wk, D, D);
    tconv(cs_wv, t_cs_wv, D, D); tconv(cs_wo, t_cs_wo, D, D);
    tconv(hp_wq, t_hp_wq, D, D); tconv(hp_wk, t_hp_wk, D, D);
    tconv(hp_wv, t_hp_wv, D, D); tconv(hp_wo, t_hp_wo, D, D);
    tconv(m1_w1, t_m1_w1, D, DFF); tconv(m1_w2, t_m1_w2, DFF, D);
    tconv(m2_w1, t_m2_w1, D, DFF); tconv(m2_w2, t_m2_w2, DFF, D);

    // ---------------- Stage 1: time-dimension Hopfield (ct) ----------------
    gemm(x,       t_ct_wq, ct_bq, bufQ, T, D, D, 0);
    gemm(x,       t_ct_wk, ct_bk, bufK, T, D, D, 0);
    gemm(bufK,    t_ct_wv, ct_bv, bufV, T, D, D, 0);       // v = proj(k)
    attn(bufQ, bufK, bufV, bufAttn, B1, 64, 64, (long long)64 * D);
    gemm(bufAttn, t_ct_wo, ct_bo, bufR1, T, D, D, 0);      // time_enc
    addln(x, bufR1, n1_g, n1_b, bufR2);                    // dim_in
    gemm(bufR2, t_m1_w1, m1_b1, bufH, T, DFF, D, 1);       // GELU fused
    gemm(bufH,  t_m1_w2, m1_b2, bufQ, T, D, DFF, 0);
    addln(bufR2, bufQ, n2_g, n2_b, bufR1);                 // dim_in (post-MLP)
    // '(b ts_d) seg d -> (b seg) ts_d d'
    permute_mid_kernel<<<dim3(BB * TSD * SEG), 256, 0, stream>>>(bufR1, bufS, TSD, SEG);

    // ---------------- Stage 2: series-dimension Hopfields ------------------
    gemm(cs_key, t_cs_wk, cs_bk, csK, FACTOR, D, D, 0);    // stored patterns
    gemm(csK,    t_cs_wv, cs_bv, csV, FACTOR, D, D, 0);
    gemm(bufS,   t_cs_wq, cs_bq, bufQ, T, D, D, 0);
    attn(bufQ, csK, csV, bufAttn, B2, 32, FACTOR, 0);      // kv stride 0: bcast
    gemm(bufAttn, t_cs_wo, cs_bo, bufA2, T, D, D, 0);      // series_h

    gemm(bufS, t_hp_wq, hp_bq, bufQ, T, D, D, 0);
    gemm(bufS, t_hp_wk, hp_bk, bufK, T, D, D, 0);
    gemm(bufK, t_hp_wv, hp_bv, bufV, T, D, D, 0);
    attn(bufQ, bufK, bufV, bufAttn, B2, 32, 32, (long long)32 * D);
    gemm(bufAttn, t_hp_wo, hp_bo, bufB2, T, D, D, 0);      // pooled_h

    addln(bufA2, bufB2, n3_g, n3_b, bufR2);                // dim_enc
    gemm(bufR2, t_m2_w1, m2_b1, bufH, T, DFF, D, 1);
    gemm(bufH,  t_m2_w2, m2_b2, bufQ, T, D, DFF, 0);
    addln(bufR2, bufQ, n4_g, n4_b, bufR1);
    // '(b seg) ts_d d -> b ts_d seg d'
    permute_mid_kernel<<<dim3(BB * TSD * SEG), 256, 0, stream>>>(bufR1, (float*)d_out, SEG, TSD);
}
